// SAM_Module_11733850652675
// MI455X (gfx1250) — matrix operand order repaired and checked by reading, not compiled, not run on hardware
//
#include <hip/hip_runtime.h>
#include <math.h>

// ---------------------------------------------------------------------------
// Problem constants (B,S,C,H,W) = (8,4,512,64,64)
// ---------------------------------------------------------------------------
constexpr int kB  = 8;
constexpr int kS  = 4;
constexpr int kC  = 512;
constexpr int kN  = 4096;            // H*W
constexpr int kCq = 128;             // C/4
constexpr int kD  = kCq * kN;        // 524288 (flattened q/k per (b,s))
constexpr int kMV = kC * kN;         // 2097152 (flattened v per (b,s))

// LDS row strides in bf16 halves: 48 halves = 96 bytes (16B aligned for b128)
constexpr int LDA = 48;
constexpr int LDB = 48;

typedef __attribute__((ext_vector_type(16))) __bf16 bf16x16;
typedef __attribute__((ext_vector_type(8)))  __bf16 bf16x8;
typedef __attribute__((ext_vector_type(2)))  __bf16 bf16x2;
typedef __attribute__((ext_vector_type(8)))  float  f32x8;
typedef __attribute__((ext_vector_type(2)))  float  f32x2;

union FragU { bf16x16 v; bf16x8 h[2]; };

// Per-thread staging registers for one 128x32 A tile + 32x128 B tile.
struct Stage {
    f32x2 a[8];            // 16 A elements as 8 contiguous pairs
    float b0[8], b1[8];    // 16 B elements as 8 (c, c+1) pairs per column n
};

// ---------------------------------------------------------------------------
// Kernel 1: fused q/k/v 1x1-conv GEMM, double-buffered LDS.
//   For (b,s):  Y(768x4096) = W_all(768x512) * X(512x4096) + bias
//   M-tiles: rows [0,128)=q, [128,256)=k, [256,768)=v.
//   BF16 WMMA 16x16x32, f32 accumulate. Block computes 128(M) x 128(N).
// ---------------------------------------------------------------------------
__global__ __launch_bounds__(256) void qkv_gemm(
    const float* __restrict__ x,
    const float* __restrict__ wq, const float* __restrict__ bq,
    const float* __restrict__ wk, const float* __restrict__ bk,
    const float* __restrict__ wv, const float* __restrict__ bv,
    float* __restrict__ qws, float* __restrict__ kws, float* __restrict__ vws)
{
    __shared__ __bf16 As[2][128 * LDA];   // A tile: [m][k], 128 x 32
    __shared__ __bf16 Bs[2][128 * LDB];   // B tile: [n][k] (transposed), 128 x 32

    const int tid  = threadIdx.x;
    const int lane = tid & 31;
    const int wv32 = tid >> 5;          // wave id 0..7
    const int g    = (lane >> 4) & 1;   // half-wave group
    const int ln   = lane & 15;

    const int bs    = blockIdx.z;       // b*4 + s
    const int s     = bs & 3;
    const int Mbase = blockIdx.y * 128; // 0..640
    const int Nbase = blockIdx.x * 128;

    // Select weight slab / bias / destination for this M-tile.
    const float* Wbase;
    const float* bias;
    float*       dst;
    int chBase = 0;
    if (Mbase == 0) {
        Wbase = wq + (size_t)s * kCq * kC; bias = bq + s * kCq;
        dst = qws + (size_t)bs * kCq * kN;
    } else if (Mbase == 128) {
        Wbase = wk + (size_t)s * kCq * kC; bias = bk + s * kCq;
        dst = kws + (size_t)bs * kCq * kN;
    } else {
        Wbase = wv + (size_t)s * kC * kC + (size_t)(Mbase - 256) * kC;
        bias  = bv + s * kC;
        dst   = vws + (size_t)bs * kC * kN;
        chBase = Mbase - 256;
    }

    const float* xb = x + (size_t)bs * kC * kN + Nbase;

    f32x8 acc[8] = {};                   // 4(M) x 2(N) tiles of 16x16
    const int wy = wv32 >> 2;            // 0..1  -> 64 M rows each
    const int wx = wv32 & 3;             // 0..3  -> 32 N cols each

    // ---- staging helpers --------------------------------------------------
    auto loadStage = [&](Stage& st, int kk) {
        const int K0 = kk * 32;
        #pragma unroll
        for (int i = 0; i < 8; ++i) {
            const int p = tid + 256 * i;           // pair index 0..2047
            // A: row = p>>4, col pair = (p&15)*2 (contiguous in global)
            const int ar = p >> 4, ac = (p & 15) * 2;
            st.a[i] = *(const f32x2*)&Wbase[(size_t)ar * kC + K0 + ac];
            // B: column n = p&127, channel pair c = (p>>7)*2 (stride kN)
            const int n = p & 127, c = (p >> 7) * 2;
            st.b0[i] = xb[(size_t)(K0 + c) * kN + n];
            st.b1[i] = xb[(size_t)(K0 + c + 1) * kN + n];
        }
    };
    auto storeStage = [&](const Stage& st, int buf) {
        #pragma unroll
        for (int i = 0; i < 8; ++i) {
            const int p = tid + 256 * i;
            const int ar = p >> 4, ac = (p & 15) * 2;
            bf16x2 av = { (__bf16)st.a[i].x, (__bf16)st.a[i].y };
            *(bf16x2*)&As[buf][ar * LDA + ac] = av;
            const int n = p & 127, c = (p >> 7) * 2;
            bf16x2 bv2 = { (__bf16)st.b0[i], (__bf16)st.b1[i] };
            *(bf16x2*)&Bs[buf][n * LDB + c] = bv2;
        }
    };

    // ---- prologue: fill buffer 0 -----------------------------------------
    Stage st;
    loadStage(st, 0);
    storeStage(st, 0);
    __syncthreads();

    // ---- main loop: compute buf (kk&1) while staging buf ((kk+1)&1) ------
    for (int kk = 0; kk < 16; ++kk) {
        const int cur = kk & 1;

        if (kk < 15) loadStage(st, kk + 1);          // global loads issue early
        if (kk < 14)
            __builtin_prefetch(xb + (size_t)(kk * 32 + 64) * kN + (tid & 127), 0, 1);

        // ---- fragments (per ISA 16-bit A/B VGPR layouts) ----
        bf16x16 af[4];
        #pragma unroll
        for (int mt = 0; mt < 4; ++mt) {
            const __bf16* p = &As[cur][(wy * 64 + mt * 16 + ln) * LDA + g * 8];
            FragU u;
            u.h[0] = *(const bf16x8*)p;          // K = g*8 .. g*8+7
            u.h[1] = *(const bf16x8*)(p + 16);   // K = 16+g*8 ..
            af[mt] = u.v;
        }
        bf16x16 bfr[2];
        #pragma unroll
        for (int nt = 0; nt < 2; ++nt) {
            const __bf16* p = &Bs[cur][(wx * 32 + nt * 16 + ln) * LDB + g * 8];
            FragU u;
            u.h[0] = *(const bf16x8*)p;          // K = g*8 .. g*8+7
            u.h[1] = *(const bf16x8*)(p + 16);   // K = 16+g*8 .. +7
            bfr[nt] = u.v;
        }

        #pragma unroll
        for (int mt = 0; mt < 4; ++mt)
            #pragma unroll
            for (int nt = 0; nt < 2; ++nt)
                acc[mt * 2 + nt] = __builtin_amdgcn_wmma_f32_16x16x32_bf16(
                    false, af[mt], false, bfr[nt],
                    (short)0, acc[mt * 2 + nt], false, false);

        if (kk < 15) storeStage(st, (kk + 1) & 1);   // fill the other buffer
        __syncthreads();
    }

    // ---- epilogue: add bias, store f32 (C/D layout: M = r + 8g, N = ln) ----
    #pragma unroll
    for (int mt = 0; mt < 4; ++mt) {
        #pragma unroll
        for (int nt = 0; nt < 2; ++nt) {
            const int ncol = Nbase + wx * 32 + nt * 16 + ln;
            f32x8 a = acc[mt * 2 + nt];
            #pragma unroll
            for (int r = 0; r < 8; ++r) {
                int ch = chBase + wy * 64 + mt * 16 + g * 8 + r;
                dst[(size_t)ch * kN + ncol] = a[r] + bias[ch];
            }
        }
    }
}

// ---------------------------------------------------------------------------
// Kernel 2: zero the reduction scratch (nq[32], nk[32], dot[128]).
// ---------------------------------------------------------------------------
__global__ void zero_red(float* __restrict__ red)
{
    if (threadIdx.x < 192) red[threadIdx.x] = 0.0f;
}

// ---------------------------------------------------------------------------
// Kernel 3: norms + all-pairs dots over D=524288 per batch.
//   red[0..31]   = sum q^2   (b*4+s)
//   red[32..63]  = sum k^2   (b*4+t)
//   red[64..191] = dot(q_s,k_t)  (b*16 + s*4 + t)
// ---------------------------------------------------------------------------
__global__ __launch_bounds__(256) void qk_reduce(
    const float* __restrict__ qws, const float* __restrict__ kws,
    float* __restrict__ red)
{
    const int b    = blockIdx.y;
    const int CL   = kD / 128;                 // 4096 elements per chunk
    const int base = blockIdx.x * CL;
    const float* qb = qws + (size_t)b * kS * kD;
    const float* kb = kws + (size_t)b * kS * kD;

    float nq[4] = {}, nk[4] = {}, dt[16] = {};
    for (int i = threadIdx.x; i < CL; i += 256) {
        const int idx = base + i;
        float qv[4], kv[4];
        #pragma unroll
        for (int t = 0; t < 4; ++t) {
            qv[t] = qb[(size_t)t * kD + idx];
            kv[t] = kb[(size_t)t * kD + idx];
        }
        #pragma unroll
        for (int t = 0; t < 4; ++t) {
            nq[t] += qv[t] * qv[t];
            nk[t] += kv[t] * kv[t];
        }
        #pragma unroll
        for (int si = 0; si < 4; ++si)
            #pragma unroll
            for (int t = 0; t < 4; ++t)
                dt[si * 4 + t] += qv[si] * kv[t];
    }

    const int lane = threadIdx.x & 31;
    #pragma unroll
    for (int t = 0; t < 4; ++t) {
        float v = nq[t];
        for (int o = 16; o > 0; o >>= 1) v += __shfl_down(v, o, 32);
        if (lane == 0) atomicAdd(&red[b * 4 + t], v);
        v = nk[t];
        for (int o = 16; o > 0; o >>= 1) v += __shfl_down(v, o, 32);
        if (lane == 0) atomicAdd(&red[32 + b * 4 + t], v);
    }
    #pragma unroll
    for (int j = 0; j < 16; ++j) {
        float v = dt[j];
        for (int o = 16; o > 0; o >>= 1) v += __shfl_down(v, o, 32);
        if (lane == 0) atomicAdd(&red[64 + b * 16 + j], v);
    }
}

// ---------------------------------------------------------------------------
// Kernel 4: cosine energy + softmax -> att (B,S,S).
// ---------------------------------------------------------------------------
__global__ void attn_softmax(const float* __restrict__ red,
                             float* __restrict__ att)
{
    const int t = threadIdx.x;
    if (t >= 32) return;
    const int b = t >> 2, s = t & 3;
    const float qn = sqrtf(red[b * 4 + s]) + 1e-10f;
    float e[4];
    #pragma unroll
    for (int j = 0; j < 4; ++j) {
        const float kn = sqrtf(red[32 + b * 4 + j]) + 1e-10f;
        e[j] = red[64 + b * 16 + s * 4 + j] / (qn * kn);
    }
    float m = e[0];
    #pragma unroll
    for (int j = 1; j < 4; ++j) m = fmaxf(m, e[j]);
    float sum = 0.f;
    #pragma unroll
    for (int j = 0; j < 4; ++j) { e[j] = expf(e[j] - m); sum += e[j]; }
    const float inv = 1.0f / sum;
    #pragma unroll
    for (int j = 0; j < 4; ++j) att[b * 16 + s * 4 + j] = e[j] * inv;
}

// ---------------------------------------------------------------------------
// Kernel 5: out[b,s] = gamma * sum_t att[b,s,t] * v[b,t] + x[b,s].
// Reads v and x exactly once each.
// ---------------------------------------------------------------------------
__global__ __launch_bounds__(256) void out_apply(
    const float* __restrict__ x, const float* __restrict__ vws,
    const float* __restrict__ att, const float* __restrict__ gamma,
    float* __restrict__ out)
{
    const int b = blockIdx.y;
    const size_t i = (size_t)blockIdx.x * 256 + threadIdx.x;   // < kMV
    const int c = (int)(i >> 12);
    const int n = (int)(i & 4095);
    const float gm = gamma[0];

    float a[16];
    #pragma unroll
    for (int j = 0; j < 16; ++j) a[j] = att[b * 16 + j];

    float vv[4];
    #pragma unroll
    for (int t = 0; t < 4; ++t)
        vv[t] = vws[(((size_t)(b * 4 + t)) * kC + c) * kN + n];

    #pragma unroll
    for (int s = 0; s < 4; ++s) {
        const size_t off = (((size_t)(b * 4 + s)) * kC + c) * kN + n;
        const float mix = a[s * 4 + 0] * vv[0] + a[s * 4 + 1] * vv[1]
                        + a[s * 4 + 2] * vv[2] + a[s * 4 + 3] * vv[3];
        out[off] = gm * mix + x[off];
    }
}

// ---------------------------------------------------------------------------
// Host-side launcher.
// ---------------------------------------------------------------------------
extern "C" void kernel_launch(void* const* d_in, const int* in_sizes, int n_in,
                              void* d_out, int out_size, void* d_ws, size_t ws_size,
                              hipStream_t stream)
{
    (void)in_sizes; (void)n_in; (void)out_size; (void)ws_size;
    const float* x     = (const float*)d_in[0];
    const float* wq    = (const float*)d_in[1];
    const float* bq    = (const float*)d_in[2];
    const float* wk    = (const float*)d_in[3];
    const float* bk    = (const float*)d_in[4];
    const float* wvp   = (const float*)d_in[5];
    const float* bvp   = (const float*)d_in[6];
    const float* gamma = (const float*)d_in[7];
    float* out = (float*)d_out;

    // Workspace layout (floats): q | k | v | red(192) | att(128)
    float* qws = (float*)d_ws;
    float* kws = qws + (size_t)kB * kS * kCq * kN;   // +16,777,216
    float* vws = kws + (size_t)kB * kS * kCq * kN;   // +16,777,216
    float* red = vws + (size_t)kB * kS * kC  * kN;   // +67,108,864
    float* att = red + 192;

    zero_red<<<1, 256, 0, stream>>>(red);
    qkv_gemm<<<dim3(kN / 128, 768 / 128, kB * kS), 256, 0, stream>>>(
        x, wq, bq, wk, bk, wvp, bvp, qws, kws, vws);
    qk_reduce<<<dim3(128, kB), 256, 0, stream>>>(qws, kws, red);
    attn_softmax<<<1, 32, 0, stream>>>(red, att);
    out_apply<<<dim3(kMV / 256, kB), 256, 0, stream>>>(x, vws, att, gamma, out);
}
